// RelativeMultiheadAttentionMechanism_51522427682851
// MI455X (gfx1250) — compile-verified, matrix operand order checked by reading, not run on hardware
//
#include <hip/hip_runtime.h>
#include <hip/hip_bf16.h>
#include <math.h>

// ---------------------------------------------------------------------------
// TransformerXL relative multi-head attention for MI455X (gfx1250, wave32).
//
// Roofline: ~78 GFLOP total, 264MB of mandatory output traffic. At 23.3 TB/s
// the memory floor is ~12us; bf16 WMMA (16x16x32, f32 accum) matches that
// (~16us at ~5 PFLOPS dense bf16), while f32 WMMA (16x16x4) would be ~10x
// compute-bound. => convert once to bf16, do all GEMMs with
// v_wmma_f32_16x16x32_bf16, keep softmax in f32.
// ---------------------------------------------------------------------------

typedef __bf16 bf16;
typedef __attribute__((ext_vector_type(16))) __bf16 v16bf;
typedef __attribute__((ext_vector_type(8)))  float  v8f;

// --- safe scalar conversions (avoid relying on compiler __bf16 casts) ------
__device__ __forceinline__ bf16 to_bf16(float x) {
  union { float f; unsigned u; } a; a.f = x;
  unsigned r = a.u + 0x7FFFu + ((a.u >> 16) & 1u);   // round-to-nearest-even
  union { unsigned short s; bf16 b; } c; c.s = (unsigned short)(r >> 16);
  return c.b;
}
__device__ __forceinline__ float bf16_to_f32(bf16 b) {
  union { unsigned short s; bf16 b; } c; c.b = b;
  union { unsigned u; float f; } a; a.u = ((unsigned)c.s) << 16;
  return a.f;
}

// --- WMMA fragment helpers --------------------------------------------------
// 16-bit A-matrix 16x32 layout (ISA 7.12.2): lane l<16 -> row l, K 0..7 &
// 16..23 ; lane l>=16 -> row l-16, K 8..15 & 24..31. Each run is 8 contiguous
// bf16 = 16 bytes -> two uint4 global loads per fragment per lane.
// B (32x16) is loaded column-major with the mirrored layout, so for our "NT"
// GEMM (B given as N x K row-major) fragment loads are also contiguous.
union FragCast { v16bf v; uint4 q[2]; };

template<bool F32>
__device__ __forceinline__ v16bf load_frag(const void* tile, long ld, int lane) {
  const int r  = lane & 15;
  const int hi = (lane >> 4) & 1;
  FragCast f;
  if constexpr (F32) {
    const float* p = (const float*)tile + (long)r * ld + hi * 8;
#pragma unroll
    for (int e = 0; e < 8; ++e) f.v[e]     = to_bf16(p[e]);
#pragma unroll
    for (int e = 0; e < 8; ++e) f.v[8 + e] = to_bf16(p[16 + e]);
  } else {
    const bf16* p = (const bf16*)tile + (long)r * ld + hi * 8;
    f.q[0] = *(const uint4*)(p);
    f.q[1] = *(const uint4*)(p + 16);
  }
  return f.v;
}

// C/D 16x16 f32 layout: lane l -> col l&15, VGPR r -> row r + 8*(l>=16).
template<bool CB>
__device__ __forceinline__ void store_tile(void* c, long csm, long csn, int lane, v8f acc) {
  const int col = lane & 15;
  const int hi  = (lane >> 4) & 1;
#pragma unroll
  for (int r = 0; r < 8; ++r) {
    long idx = (long)(r + hi * 8) * csm + (long)col * csn;
    if constexpr (CB) ((bf16*)c)[idx] = to_bf16(acc[r]);
    else              ((float*)c)[idx] = acc[r];
  }
}

// --- generic batched NT GEMM: C[m,n] = sum_k A[m,k] * B[n,k] ---------------
// Batch z decomposes as (b = z/Hh, h = z%Hh); per-operand (b,h) strides let
// one kernel cover full-matrix projections and per-head attention GEMMs.
// Block = 256 threads = 8 wave32 arranged 4(M) x 2(N).
// Wave tile = 32 x (16*NW)  ->  block tile 128 x (32*NW).
// NW=4: 2 A-frags + 4 B-frags feed 8 WMMAs per 32-K step (1.33 wmma/frag).
// NW=2: used only for the N=64 aw@v GEMM.
template<bool AF32, bool CBF16, int NW>
__global__ __launch_bounds__(256)
void gemm_nt(const void* __restrict__ Ab, long lda, long sAb, long sAh,
             const bf16* __restrict__ Bb, long ldb, long sBb, long sBh,
             void* __restrict__ Cb, long csm, long csn, long sCb, long sCh,
             int K, int Hh)
{
  const int z = blockIdx.z, b = z / Hh, h = z - b * Hh;

  const void* A0;
  if constexpr (AF32) A0 = (const void*)((const float*)Ab + (long)b * sAb + (long)h * sAh);
  else                A0 = (const void*)((const bf16*)Ab  + (long)b * sAb + (long)h * sAh);
  const bf16* B0 = Bb + (long)b * sBb + (long)h * sBh;
  void* C0;
  if constexpr (CBF16) C0 = (void*)((bf16*)Cb  + (long)b * sCb + (long)h * sCh);
  else                 C0 = (void*)((float*)Cb + (long)b * sCb + (long)h * sCh);

  const int wave = threadIdx.x >> 5, lane = threadIdx.x & 31;
  const int row = blockIdx.y * 128 + (wave >> 1) * 32;
  const int col = blockIdx.x * (32 * NW) + (wave & 1) * (16 * NW);

  v8f acc0[NW], acc1[NW];
#pragma unroll
  for (int j = 0; j < NW; ++j) { acc0[j] = {}; acc1[j] = {}; }

  for (int k0 = 0; k0 < K; k0 += 32) {
    const void* a0p; const void* a1p;
    if constexpr (AF32) {
      a0p = (const float*)A0 + (long)row * lda + k0;
      a1p = (const float*)A0 + (long)(row + 16) * lda + k0;
    } else {
      a0p = (const bf16*)A0 + (long)row * lda + k0;
      a1p = (const bf16*)A0 + (long)(row + 16) * lda + k0;
    }

    if (k0 + 32 < K) {  // global_prefetch_b8 for the next K-step (near caches)
      __builtin_prefetch((const char*)a0p + 32 * (AF32 ? 4 : 2), 0, 3);
      __builtin_prefetch((const char*)(B0 + (long)col * ldb + k0) + 64, 0, 3);
    }

    v16bf a0 = load_frag<AF32>(a0p, lda, lane);
    v16bf a1 = load_frag<AF32>(a1p, lda, lane);
    v16bf bb[NW];
#pragma unroll
    for (int j = 0; j < NW; ++j)
      bb[j] = load_frag<false>(B0 + (long)(col + 16 * j) * ldb + k0, ldb, lane);

#pragma unroll
    for (int j = 0; j < NW; ++j) {
      acc0[j] = __builtin_amdgcn_wmma_f32_16x16x32_bf16(false, a0, false, bb[j], (short)0, acc0[j], false, false);
      acc1[j] = __builtin_amdgcn_wmma_f32_16x16x32_bf16(false, a1, false, bb[j], (short)0, acc1[j], false, false);
    }
  }

  auto cptr = [&](int mo, int no) -> void* {
    long o = (long)(row + mo) * csm + (long)(col + no) * csn;
    if constexpr (CBF16) return (void*)((bf16*)C0 + o);
    else                 return (void*)((float*)C0 + o);
  };
#pragma unroll
  for (int j = 0; j < NW; ++j) {
    store_tile<CBF16>(cptr(0,  16 * j), csm, csn, lane, acc0[j]);
    store_tile<CBF16>(cptr(16, 16 * j), csm, csn, lane, acc1[j]);
  }
}

// --- elementwise helpers ----------------------------------------------------
__global__ void cvt_f32_bf16(const float* __restrict__ s, bf16* __restrict__ d, long n) {
  long i = (long)blockIdx.x * blockDim.x + threadIdx.x;
  if (i < n) d[i] = to_bf16(s[i]);
}

__global__ void transpose_cvt(const float* __restrict__ s, bf16* __restrict__ d, int K, int N) {
  long i = (long)blockIdx.x * blockDim.x + threadIdx.x;
  if (i < (long)K * N) {
    int k = (int)(i / N);
    int n = (int)(i - (long)k * N);
    d[(long)n * K + k] = to_bf16(s[i]);   // d[n,k] = s[k,n]
  }
}

__global__ void add_uv_cvt(const float* __restrict__ q, const float* __restrict__ uv,
                           bf16* __restrict__ d, long n, int innerMask) {
  long i = (long)blockIdx.x * blockDim.x + threadIdx.x;
  if (i < n) d[i] = to_bf16(q[i] + uv[i & innerMask]);
}

// --- fused rel-shift + mask + softmax (in place on the aw output buffer) ---
// aw:    [B*H, Q, 2048] f32, pre-filled with AC by the WMMA GEMM
// bdraw: [B*H, Q, 2048] bf16, BDraw[i,p] = (q_i+v)·pos_p
// rel_shift (q=1024,k=2048): with d=j-i:
//   d<=1024 -> BDraw[i,1023+d];  d==1025 -> 0;  d>=1026 -> BDraw[i+1,d-1026]
__global__ __launch_bounds__(256)
void softmax_relshift(float* __restrict__ aw, const bf16* __restrict__ bdraw,
                      const int* __restrict__ mask)
{
  const int i = blockIdx.x;          // query row
  const int z = blockIdx.y;          // b*H + h
  const int b = z >> 4;              // H = 16
  float*       row  = aw    + ((long)z * 1024 + i) * 2048;
  const bf16*  bd0  = bdraw + ((long)z * 1024 + i) * 2048;
  const int*   mrow = mask  + ((long)b * 1024 + i) * 2048;
  const int tid = threadIdx.x, lane = tid & 31, wave = tid >> 5;

  float e[8];
  float mx = -3.4028235e38f;
#pragma unroll
  for (int t = 0; t < 8; ++t) {
    int j = tid + t * 256;
    int d = j - i;
    float bd;
    if (d <= 1024)      bd = bf16_to_f32(bd0[1023 + d]);
    else if (d == 1025) bd = 0.0f;
    else                bd = bf16_to_f32(bd0[2048 + (d - 1026)]);   // row i+1
    float ev = (row[j] + bd) * 0.125f;                              // /sqrt(64)
    if (mrow[j] == 0) ev = -3.4028235e38f;
    e[t] = ev;
    mx = fmaxf(mx, ev);
  }

  __shared__ float red[8];
  for (int off = 16; off > 0; off >>= 1) mx = fmaxf(mx, __shfl_xor(mx, off, 32));
  if (lane == 0) red[wave] = mx;
  __syncthreads();
  mx = red[0];
#pragma unroll
  for (int w = 1; w < 8; ++w) mx = fmaxf(mx, red[w]);

  float sum = 0.0f;
#pragma unroll
  for (int t = 0; t < 8; ++t) { e[t] = __expf(e[t] - mx); sum += e[t]; }
  __syncthreads();
  for (int off = 16; off > 0; off >>= 1) sum += __shfl_xor(sum, off, 32);
  if (lane == 0) red[wave] = sum;
  __syncthreads();
  sum = 0.0f;
#pragma unroll
  for (int w = 0; w < 8; ++w) sum += red[w];
  float inv = 1.0f / sum;
#pragma unroll
  for (int t = 0; t < 8; ++t) row[tid + t * 256] = e[t] * inv;
}

// ---------------------------------------------------------------------------
extern "C" void kernel_launch(void* const* d_in, const int* in_sizes, int n_in,
                              void* d_out, int out_size, void* d_ws, size_t ws_size,
                              hipStream_t stream)
{
  const float* key_x   = (const float*)d_in[0];
  const float* query   = (const float*)d_in[1];
  const float* memory  = (const float*)d_in[2];
  const float* posemb  = (const float*)d_in[3];
  const int*   mask    = (const int*)d_in[4];
  const float* u       = (const float*)d_in[5];
  const float* v       = (const float*)d_in[6];
  const float* w_key   = (const float*)d_in[7];
  const float* w_value = (const float*)d_in[8];
  const float* w_query = (const float*)d_in[9];
  const float* w_pos   = (const float*)d_in[10];
  const float* w_out   = (const float*)d_in[11];

  const long Bsz = 2, Q = 1024, Kt = 2048, D = 1024, H = 16;

  // ---- workspace layout (~198 MiB) ----
  size_t off = 0;
  char* ws = (char*)d_ws;
  auto take = [&](size_t bytes) { size_t r = off; off += (bytes + 255) & ~(size_t)255; return r; };
  bf16*  kv    = (bf16*)(ws + take((size_t)Bsz * Kt * D * 2));   // [b,2048,1024] = [memory;key_x]
  bf16*  bq    = (bf16*)(ws + take((size_t)Bsz * Q  * D * 2));
  bf16*  bpos  = (bf16*)(ws + take((size_t)Kt * D * 2));
  bf16*  wkT   = (bf16*)(ws + take((size_t)D * D * 2));
  bf16*  wvT   = (bf16*)(ws + take((size_t)D * D * 2));
  bf16*  wqT   = (bf16*)(ws + take((size_t)D * D * 2));
  bf16*  wpT   = (bf16*)(ws + take((size_t)D * D * 2));
  bf16*  woT   = (bf16*)(ws + take((size_t)D * D * 2));
  bf16*  kpr   = (bf16*)(ws + take((size_t)Bsz * Kt * D * 2));   // k heads [b,j,h*64+d]
  bf16*  valT  = (bf16*)(ws + take((size_t)Bsz * Kt * D * 2));   // [b, h*64+d, j]  (pre-transposed)
  bf16*  posp  = (bf16*)(ws + take((size_t)Kt * D * 2));         // [p, h*64+d]
  float* qf    = (float*)(ws + take((size_t)Bsz * Q * D * 4));   // q projection f32
  bf16*  qu    = (bf16*)(ws + take((size_t)Bsz * Q * D * 2));    // bf16(q+u)
  bf16*  qvb   = (bf16*)(ws + take((size_t)Bsz * Q * D * 2));    // bf16(q+v)
  bf16*  bdraw = (bf16*)(ws + take((size_t)Bsz * H * Q * Kt * 2)); // 128 MiB
  float* cv1   = (float*)(ws + take((size_t)Bsz * Q * D * 4));
  if (off > ws_size) return;   // workspace too small: bail deterministically

  float* out_cv = (float*)d_out;                // [B,Q,D]
  float* out_aw = out_cv + (size_t)Bsz * Q * D; // [B,H,Q,Kt]

  dim3 blk(256);
  auto nb = [](long n) { return dim3((unsigned)((n + 255) / 256)); };

  // 1) bf16 copies; kv = concat(memory, key_x) along sequence per batch
  cvt_f32_bf16<<<nb(Q * D), blk, 0, stream>>>(memory,         kv,             Q * D);
  cvt_f32_bf16<<<nb(Q * D), blk, 0, stream>>>(key_x,          kv + Q * D,     Q * D);
  cvt_f32_bf16<<<nb(Q * D), blk, 0, stream>>>(memory + Q * D, kv + 2 * Q * D, Q * D);
  cvt_f32_bf16<<<nb(Q * D), blk, 0, stream>>>(key_x + Q * D,  kv + 3 * Q * D, Q * D);
  cvt_f32_bf16<<<nb(Bsz * Q * D), blk, 0, stream>>>(query, bq, Bsz * Q * D);
  cvt_f32_bf16<<<nb(Kt * D), blk, 0, stream>>>(posemb, bpos, Kt * D);
  transpose_cvt<<<nb(D * D), blk, 0, stream>>>(w_key,   wkT, (int)D, (int)D);
  transpose_cvt<<<nb(D * D), blk, 0, stream>>>(w_value, wvT, (int)D, (int)D);
  transpose_cvt<<<nb(D * D), blk, 0, stream>>>(w_query, wqT, (int)D, (int)D);
  transpose_cvt<<<nb(D * D), blk, 0, stream>>>(w_pos,   wpT, (int)D, (int)D);
  transpose_cvt<<<nb(D * D), blk, 0, stream>>>(w_out,   woT, (int)D, (int)D);

  // 2) k = kv @ w_key          (M=4096, N=1024, K=1024) -> bf16 row-major
  gemm_nt<false, true, 4><<<dim3(8, 32, 1), blk, 0, stream>>>(
      kv, D, 0, 0, wkT, D, 0, 0, kpr, D, 1, 0, 0, (int)D, 1);

  // 3) valT = (kv @ w_value)^T per batch: C[m=j, n=h*64+d] stored at [n, j]
  gemm_nt<false, true, 4><<<dim3(8, 16, 2), blk, 0, stream>>>(
      kv, D, Kt * D, 0, wvT, D, 0, 0, valT, 1, Kt, Kt * D, 0, (int)D, 1);

  // 4) q = query @ w_query -> f32; then qu = bf16(q+u), qv = bf16(q+v)
  gemm_nt<false, false, 4><<<dim3(8, 16, 1), blk, 0, stream>>>(
      bq, D, 0, 0, wqT, D, 0, 0, qf, D, 1, 0, 0, (int)D, 1);
  add_uv_cvt<<<nb(Bsz * Q * D), blk, 0, stream>>>(qf, u, qu,  Bsz * Q * D, 1023);
  add_uv_cvt<<<nb(Bsz * Q * D), blk, 0, stream>>>(qf, v, qvb, Bsz * Q * D, 1023);

  // 5) pos projection (M=2048, N=1024, K=1024) -> bf16 row-major
  gemm_nt<false, true, 4><<<dim3(8, 16, 1), blk, 0, stream>>>(
      bpos, D, 0, 0, wpT, D, 0, 0, posp, D, 1, 0, 0, (int)D, 1);

  // 6) AC[z,i,j] = (q+u)·k  straight into the aw output buffer (f32)
  //    per (b,h): A = qu[b,:,h*64:...], B = kpr[b,:,h*64:...]  (NT, K=64)
  gemm_nt<false, false, 4><<<dim3(16, 8, 32), blk, 0, stream>>>(
      qu,  D, Q * D,  64, kpr, D, Kt * D, 64,
      out_aw, Kt, 1, H * Q * Kt, Q * Kt, 64, (int)H);

  // 7) BDraw[z,i,p] = (q+v)·pos  (bf16, gathered with rel-shift in softmax)
  gemm_nt<false, true, 4><<<dim3(16, 8, 32), blk, 0, stream>>>(
      qvb, D, Q * D, 64, posp, D, 0, 64,
      bdraw, Kt, 1, H * Q * Kt, Q * Kt, 64, (int)H);

  // 8) e = (AC + rel_shift(BD))/8, mask, softmax -> aw in place
  softmax_relshift<<<dim3((unsigned)Q, (unsigned)(Bsz * H)), blk, 0, stream>>>(
      out_aw, bdraw, mask);

  // 9) cv1[b,i,h*64+d] = sum_j aw[z,i,j] * val[b,j,h,d]
  //    A = aw (f32, converted in-register), B = valT rows (NT, K=2048), N=64
  gemm_nt<true, false, 2><<<dim3(1, 8, 32), blk, 0, stream>>>(
      out_aw, Kt, H * Q * Kt, Q * Kt, valT, Kt, Kt * D, 64 * Kt,
      cv1, D, 1, Q * D, 64, (int)Kt, (int)H);

  // 10) out_cv = cv1 @ w_out   (M=2048, N=1024, K=1024), f32 A and C
  gemm_nt<true, false, 4><<<dim3(8, 16, 1), blk, 0, stream>>>(
      cv1, D, 0, 0, woT, D, 0, 0, out_cv, D, 1, 0, 0, (int)D, 1);
}
